// Autoformer_64879775973915
// MI455X (gfx1250) — matrix operand order repaired and checked by reading, not compiled, not run on hardware
//
#include <hip/hip_runtime.h>

// ---------------- model constants ----------------
#define BK   256      // batch
#define LE   96       // encoder seq len (PRED_LEN)
#define LD   144      // decoder seq len (LABEL_LEN + PRED_LEN)
#define DM   256      // d_model
#define DFF  512
#define CIN  64       // enc_in / c_out / out_size
#define TOPK 4        // int(log(96)) == int(log(144)) == 4

typedef __bf16 v16bf __attribute__((ext_vector_type(16)));
typedef float  v8f   __attribute__((ext_vector_type(8)));
typedef unsigned int u32x4 __attribute__((ext_vector_type(4)));
typedef int          i32x4 __attribute__((ext_vector_type(4)));
typedef int          i32x8 __attribute__((ext_vector_type(8)));

#if defined(__has_builtin)
#if __has_builtin(__builtin_amdgcn_tensor_load_to_lds)
#define HAVE_TDM 1
#endif
#endif
#ifndef HAVE_TDM
#define HAVE_TDM 0
#endif

__device__ __forceinline__ unsigned short f2bfbits(float x) {
    union { float f; unsigned u; } a; a.f = x;
    unsigned r = (a.u + 0x7FFFu + ((a.u >> 16) & 1u)) >> 16;   // RNE
    return (unsigned short)r;
}

// fp32 -> packed bf16 conversion (2 elements / thread, dword stores)
__global__ void cvt_k(const float* __restrict__ x, unsigned* __restrict__ y, long n2) {
    long i = blockIdx.x * (long)blockDim.x + threadIdx.x;
    if (i >= n2) return;
    unsigned lo = f2bfbits(x[2 * i]);
    unsigned hi = f2bfbits(x[2 * i + 1]);
    y[i] = lo | (hi << 16);
}

// ---------------- WMMA GEMM -------------------------------------------------
// C[M,N] = op(Abf16[M,K] @ Bbf16) with bias / residual / accumulate / GELU.
// A panel (128 x K) staged to LDS by the Tensor Data Mover (row-padded),
// B panel (K x 16) staged column-major (padded).  Batched via blockIdx.z.
enum { GF_BIAS = 1, GF_RESID = 2, GF_ACCUM = 4, GF_GELU = 8, GF_BTRANS = 16 };

template<int F>
__launch_bounds__(256)
__global__ void gemm_k(const unsigned short* __restrict__ A, const unsigned short* __restrict__ Bm,
                       const float* __restrict__ bias, const float* __restrict__ R,
                       float* __restrict__ C, int M, int N, int K,
                       long sA, long sB, long sC)
{
    extern __shared__ unsigned short smem[];
    const int Kp = K + 8;                       // +16B per row: bank rotation, keeps 16B align
    unsigned short* As = smem;                  // 128 rows x Kp (row-major, padded)
    unsigned short* Bs = smem + 128 * Kp;       // 16 cols x Kp (column-major, padded)
    const int tid    = threadIdx.x;
    const int wave   = tid >> 5;
    const int lane   = tid & 31;
    const int tileM0 = blockIdx.x * 128;
    const int tileM  = tileM0 + wave * 16;
    const int tileN  = blockIdx.y * 16;
    const long aOff  = (long)blockIdx.z * sA;
    const long bOff  = (long)blockIdx.z * sB;
    const long cOff  = (long)blockIdx.z * sC;
    const int Mrem   = M - tileM0;              // >= 1

    // ---- stage A panel via TDM (async tensor DMA), manual fallback ----
#if HAVE_TDM
    if (wave == 0) {
        unsigned long long ga = (unsigned long long)(const void*)(A + aOff + (long)tileM0 * K);
        unsigned ldsa = (unsigned)(unsigned long long)(const void*)As;
        // pad_interval: K/2 dwords per row -> code log2(K)-2 (K=64->4, 256->6, 512->7)
        unsigned padc = (unsigned)(31 - __builtin_clz((unsigned)K)) - 2u;
        u32x4 g0 = { 1u,                                    // count=1, user descriptor
                     ldsa,                                  // lds_addr
                     (unsigned)(ga & 0xffffffffull),
                     (unsigned)((ga >> 32) & 0x1ffffffull) | (2u << 30) };  // type=2 (image)
        i32x8 g1 = { (int)((1u << 16) | (1u << 20) | (padc << 22) | (3u << 25)), // 2B elems, pad_en, interval, 4-dword pad
                     (int)((K & 0xffff) << 16),                                  // tensor_dim0 lo
                     (int)(((unsigned)K >> 16) | (((unsigned)Mrem & 0xffff) << 16)), // dim0 hi | tensor_dim1 lo
                     (int)((((unsigned)Mrem >> 16) & 0xffff) | (((unsigned)K & 0xffff) << 16)), // dim1 hi | tile_dim0=K
                     128,                                   // tile_dim1 = 128 rows (OOB rows -> zero)
                     K,                                     // tensor_dim0_stride = K (contiguous rows)
                     0, 0 };
        i32x4 gz = { 0, 0, 0, 0 };
#if __clang_major__ >= 23
        i32x8 gz8 = { 0, 0, 0, 0, 0, 0, 0, 0 };
        __builtin_amdgcn_tensor_load_to_lds(g0, g1, gz, gz, gz8, 0);
#else
        __builtin_amdgcn_tensor_load_to_lds(g0, g1, gz, gz, 0);
#endif
        __builtin_amdgcn_s_wait_tensorcnt(0);
    }
#else
    {
        const int rowv = K >> 3;                // uint4 chunks per row
        for (int i = tid; i < 128 * rowv; i += 256) {
            int row = i / rowv, kc = (i % rowv) * 8;
            uint4 v = { 0u, 0u, 0u, 0u };
            if (row < Mrem)
                v = *(const uint4*)(A + aOff + (long)(tileM0 + row) * K + kc);
            *(uint4*)(As + row * Kp + kc) = v;
        }
    }
#endif

    // ---- stage B panel column-major ----
    if (F & GF_BTRANS) {                        // B given as [N,K]: rows contiguous along K
        const int rowv = K >> 3;
        for (int i = tid; i < 16 * rowv; i += 256) {
            int c = i / rowv, kc = (i % rowv) * 8;
            uint4 v = *(const uint4*)(Bm + bOff + (long)(tileN + c) * K + kc);
            *(uint4*)(Bs + c * Kp + kc) = v;
        }
    } else {                                    // B given as [K,N]
        for (int i = tid; i < K * 8; i += 256) {
            int k = i >> 3, cp = (i & 7) * 2;
            unsigned v = *(const unsigned*)(Bm + bOff + (long)k * N + tileN + cp);
            Bs[cp * Kp + k]       = (unsigned short)v;
            Bs[(cp + 1) * Kp + k] = (unsigned short)(v >> 16);
        }
    }
    __syncthreads();

    const bool act  = (tileM < M);              // wave-uniform: EXEC all-1s inside
    const int  half = lane >> 4;
    const int  lr   = lane & 15;
    v8f acc = { 0.f, 0.f, 0.f, 0.f, 0.f, 0.f, 0.f, 0.f };

    if (act) {
        // A frag: lane half 0 -> K 0..7 / 16..23 ; half 1 -> 8..15 / 24..31
        const unsigned short* arow = As + (wave * 16 + lr) * Kp + half * 8;
        // B frag: lane half 0 -> K 0..7 / 16..23 ; half 1 -> 8..15 / 24..31, column lr
        const unsigned short* bcol = Bs + lr * Kp + half * 8;
        for (int kk = 0; kk < K; kk += 32) {
            union { uint4 q[2]; v16bf v; } ua, ub;
            ua.q[0] = *(const uint4*)(arow + kk);          // ds_load_b128
            ua.q[1] = *(const uint4*)(arow + kk + 16);
            ub.q[0] = *(const uint4*)(bcol + kk);
            ub.q[1] = *(const uint4*)(bcol + kk + 16);
            acc = __builtin_amdgcn_wmma_f32_16x16x32_bf16(
                    false, ua.v, false, ub.v, (short)0, acc, false, false);
        }
        const int rbase = half * 8;             // C/D: VGPR r -> M = r / 8+r
        #pragma unroll
        for (int r = 0; r < 8; ++r) {
            const long row = tileM + rbase + r;
            const long col = tileN + lr;
            float v = acc[r];
            if (F & GF_BIAS)  v += bias[col];
            if (F & GF_RESID) v += R[cOff + row * N + col];
            if (F & GF_GELU)  v = 0.5f * v * (1.0f + erff(v * 0.70710678118f));
            if (F & GF_ACCUM) C[cOff + row * N + col] += v;
            else              C[cOff + row * N + col] = v;
        }
    }
}

// ---------------- elementwise / reduction kernels ---------------------------
__global__ void zero_k(float* p, long n) {
    long i = blockIdx.x * (long)blockDim.x + threadIdx.x;
    if (i < n) p[i] = 0.f;
}
__global__ void add2_k(const float* a, const float* b, float* c, long n) {
    long i = blockIdx.x * (long)blockDim.x + threadIdx.x;
    if (i < n) c[i] = a[i] + b[i];
}
__global__ void shift_k(const float* __restrict__ x, float* __restrict__ y,
                        int B, int L, int C, int sh) {
    long i = blockIdx.x * (long)blockDim.x + threadIdx.x;
    long tot = (long)B * L * C;
    if (i >= tot) return;
    int c = (int)(i % C); long r = i / C; int t = (int)(r % L); int b = (int)(r / L);
    int ts = (t + sh + L) % L;
    y[i] = x[((long)b * L + ts) * C + c];
}
__global__ void decomp_k(const float* __restrict__ x, float* __restrict__ seas,
                         float* __restrict__ trend, int B, int L, int C, int tmode) {
    long i = blockIdx.x * (long)blockDim.x + threadIdx.x;
    long tot = (long)B * L * C;
    if (i >= tot) return;
    int c = (int)(i % C); long r = i / C; int t = (int)(r % L); int b = (int)(r / L);
    const float* xb = x + (long)b * L * C + c;
    float s = 0.f;
    for (int j = -12; j <= 12; ++j) {
        int tt = t + j; tt = tt < 0 ? 0 : (tt >= L ? L - 1 : tt);
        s += xb[(long)tt * C];
    }
    float mm = s * (1.0f / 25.0f);
    seas[i] = x[i] - mm;
    if (tmode == 1) trend[i] = mm;
    else if (tmode == 2) trend[i] += mm;
}
__global__ void meanT_k(const float* __restrict__ x, float* __restrict__ m,
                        int B, int L, int C) {
    long i = blockIdx.x * (long)blockDim.x + threadIdx.x;
    if (i >= (long)B * C) return;
    int c = (int)(i % C), b = (int)(i / C);
    const float* p = x + (long)b * L * C + c;
    float s = 0.f;
    for (int t = 0; t < L; ++t) s += p[(long)t * C];
    m[i] = s / L;
}
__global__ void build_trend_k(const float* __restrict__ txe, const float* __restrict__ mb,
                              float* __restrict__ tr) {
    long i = blockIdx.x * (long)blockDim.x + threadIdx.x;
    long tot = (long)BK * LD * CIN;
    if (i >= tot) return;
    int c = (int)(i % CIN); long r = i / CIN; int t = (int)(r % LD); int b = (int)(r / LD);
    tr[i] = (t < 48) ? txe[((long)b * LE + 48 + t) * CIN + c] : mb[b * CIN + c];
}
__global__ void build_seas_k(const float* __restrict__ sxe, float* __restrict__ si) {
    long i = blockIdx.x * (long)blockDim.x + threadIdx.x;
    long tot = (long)BK * LD * CIN;
    if (i >= tot) return;
    int c = (int)(i % CIN); long r = i / CIN; int t = (int)(r % LD); int b = (int)(r / LD);
    si[i] = (t < 48) ? sxe[((long)b * LE + 48 + t) * CIN + c] : 0.f;
}
__global__ void ln_k(const float* __restrict__ x, float* __restrict__ y,
                     const float* __restrict__ g, const float* __restrict__ be, long ntok) {
    long t = blockIdx.x * (long)blockDim.x + threadIdx.x;
    if (t >= ntok) return;
    const float* xr = x + t * DM;
    float s = 0.f, ss = 0.f;
    for (int c = 0; c < DM; ++c) { float v = xr[c]; s += v; ss += v * v; }
    float mu = s * (1.0f / DM);
    float inv = rsqrtf(ss * (1.0f / DM) - mu * mu + 1e-5f);
    float* yr = y + t * DM;
    for (int c = 0; c < DM; ++c) yr[c] = (xr[c] - mu) * inv * g[c] + be[c];
}
__global__ void subTmean_k(float* __restrict__ y, int B, int L) {
    long i = blockIdx.x * (long)blockDim.x + threadIdx.x;
    if (i >= (long)B * DM) return;
    int c = (int)(i % DM), b = (int)(i / DM);
    float* p = y + (long)b * L * DM + c;
    float s = 0.f;
    for (int t = 0; t < L; ++t) s += p[(long)t * DM];
    float m = s / L;
    for (int t = 0; t < L; ++t) p[(long)t * DM] -= m;
}
__global__ void diagmean_k(const float* __restrict__ S, float* __restrict__ mv,
                           int B, int L) {
    long i = blockIdx.x * (long)blockDim.x + threadIdx.x;
    if (i >= (long)B * L) return;
    int tau = (int)(i % L), b = (int)(i / L);
    const float* Sb = S + (long)b * L * L;
    float s = 0.f;
    for (int t = 0; t < L; ++t) s += Sb[(long)((t + tau) % L) * L + t];
    mv[i] = s * (1.0f / 256.0f);
}
__global__ void topk_k(const float* __restrict__ mv, float* __restrict__ w,
                       int* __restrict__ dly, int B, int L) {
    int b = blockIdx.x * blockDim.x + threadIdx.x;
    if (b >= B) return;
    float bv[TOPK] = { -1e30f, -1e30f, -1e30f, -1e30f };
    int   bi[TOPK] = { 0, 0, 0, 0 };
    for (int t = 0; t < L; ++t) {
        float v = mv[b * L + t];
        if (v > bv[TOPK - 1]) {
            int p = TOPK - 1;
            while (p > 0 && v > bv[p - 1]) { bv[p] = bv[p - 1]; bi[p] = bi[p - 1]; --p; }
            bv[p] = v; bi[p] = t;
        }
    }
    float m = bv[0], s = 0.f, e[TOPK];
    for (int j = 0; j < TOPK; ++j) { e[j] = __expf(bv[j] - m); s += e[j]; }
    for (int j = 0; j < TOPK; ++j) { w[b * TOPK + j] = e[j] / s; dly[b * TOPK + j] = bi[j]; }
}
__global__ void agg_k(const float* __restrict__ v, const float* __restrict__ w,
                      const int* __restrict__ dly, float* __restrict__ o, int B, int L) {
    long i = blockIdx.x * (long)blockDim.x + threadIdx.x;
    long tot = (long)B * L * DM;
    if (i >= tot) return;
    int c = (int)(i % DM); long r = i / DM; int l = (int)(r % L); int b = (int)(r / L);
    float s = 0.f;
    for (int j = 0; j < TOPK; ++j) {
        int d = dly[b * TOPK + j];
        s += w[b * TOPK + j] * v[((long)b * L + (l + d) % L) * DM + c];
    }
    o[i] = s;
}
__global__ void slice_k(const float* __restrict__ of, float* __restrict__ out) {
    long i = blockIdx.x * (long)blockDim.x + threadIdx.x;
    long tot = (long)BK * LE * CIN;
    if (i >= tot) return;
    int c = (int)(i % CIN); long r = i / CIN; int t = (int)(r % LE); int b = (int)(r / LE);
    out[i] = of[((long)b * LD + 48 + t) * CIN + c];
}

// ---------------- host orchestration ----------------------------------------
static inline dim3 eg(long n) { return dim3((unsigned)((n + 255) / 256)); }

static void launch_gemm(int flags, const unsigned short* A, const unsigned short* Bm,
                        const float* bias, const float* R, float* C, int M, int N, int K,
                        long sA, long sB, long sC, int batch, hipStream_t s)
{
    dim3 g((M + 127) / 128, N / 16, batch), b(256);
    size_t sm = (size_t)144 * (K + 8) * 2;     // A(128 rows) + B(16 cols), padded, bf16
#define GEMM_CASE(FL) case FL: gemm_k<FL><<<g, b, sm, s>>>(A, Bm, bias, R, C, M, N, K, sA, sB, sC); break;
    switch (flags) {
        GEMM_CASE(0)
        GEMM_CASE(GF_BIAS)
        GEMM_CASE(GF_ACCUM)
        GEMM_CASE(GF_RESID)
        GEMM_CASE(GF_GELU)
        GEMM_CASE(GF_BIAS | GF_RESID)
        GEMM_CASE(GF_BTRANS)
        default: break;
    }
#undef GEMM_CASE
}

extern "C" void kernel_launch(void* const* d_in, const int* in_sizes, int n_in,
                              void* d_out, int out_size, void* d_ws, size_t ws_size,
                              hipStream_t stream)
{
    (void)in_sizes; (void)n_in; (void)out_size; (void)ws_size;
    auto P = [&](int i) -> const float* { return (const float*)d_in[i]; };

    const float* xe = P(0);                 // [BK,96,64]
    const long ME = (long)BK * LE, MD = (long)BK * LD;

    float* ws = (float*)d_ws;
    long off = 0;
    auto alloc = [&](long n) { float* p = ws + off; off += n; return p; };
    float* enc   = alloc(ME * DM);
    float* dec   = alloc(MD * DM);
    float* qb    = alloc(MD * DM);
    float* kb    = alloc(MD * DM);
    float* vb    = alloc(MD * DM);
    float* big   = alloc(MD * DFF);         // FF hidden / attention scores
    float* ta    = alloc(MD * DM);
    float* tb    = alloc(MD * DM);
    float* tt    = alloc(MD * DM);          // t1+t2+t3 trend accumulator
    float* trend = alloc(MD * CIN);
    float* sxe   = alloc(ME * CIN);
    float* txe   = alloc(ME * CIN);
    float* meanb = alloc((long)BK * CIN);
    float* seasi = alloc(MD * CIN);
    float* mv    = alloc((long)BK * LD);
    float* wgt   = alloc((long)BK * TOPK);
    int*   dly   = (int*)alloc((long)BK * TOPK);
    float* s1    = alloc(MD * CIN);
    float* ofb   = alloc(MD * CIN);
    unsigned short* abf = (unsigned short*)alloc(MD * DFF / 2);   // bf16 A mirror
    unsigned short* bbf = (unsigned short*)alloc(MD * DM / 2);    // bf16 B mirror

    auto cvt = [&](const float* x, unsigned short* y, long n) {
        cvt_k<<<eg(n / 2), 256, 0, stream>>>(x, (unsigned*)y, n / 2);
    };

    auto attn_block = [&](const float* xq, int Lq, const float* xkv, int Lkv,
                          int pb, float* dst) {
        const float *Wq = P(pb + 0), *bq = P(pb + 1), *Wk = P(pb + 2), *bk2 = P(pb + 3),
                    *Wv = P(pb + 4), *bv2 = P(pb + 5), *Wo = P(pb + 6), *bo = P(pb + 7);
        int Mq = BK * Lq;
        cvt(xq, abf, (long)Mq * DM);
        cvt(Wq, bbf, (long)DM * DM);
        launch_gemm(GF_BIAS, abf, bbf, bq, nullptr, qb, Mq, DM, DM, 0, 0, 0, 1, stream);
        if (Lkv == Lq) {
            cvt(Wk, bbf, (long)DM * DM);
            launch_gemm(GF_BIAS, abf, bbf, bk2, nullptr, kb, Mq, DM, DM, 0, 0, 0, 1, stream);
            cvt(Wv, bbf, (long)DM * DM);
            launch_gemm(GF_BIAS, abf, bbf, bv2, nullptr, vb, Mq, DM, DM, 0, 0, 0, 1, stream);
        } else {  // cross attention: zero-pad K/V along time to Lq
            zero_k<<<eg((long)Mq * DM), 256, 0, stream>>>(kb, (long)Mq * DM);
            zero_k<<<eg((long)Mq * DM), 256, 0, stream>>>(vb, (long)Mq * DM);
            cvt(xkv, abf, (long)BK * Lkv * DM);
            cvt(Wk, bbf, (long)DM * DM);
            launch_gemm(GF_BIAS, abf, bbf, bk2, nullptr, kb, Lkv, DM, DM,
                        (long)Lkv * DM, 0, (long)Lq * DM, BK, stream);
            cvt(Wv, bbf, (long)DM * DM);
            launch_gemm(GF_BIAS, abf, bbf, bv2, nullptr, vb, Lkv, DM, DM,
                        (long)Lkv * DM, 0, (long)Lq * DM, BK, stream);
        }
        // S = Q K^T per batch (WMMA), circular-correlation mean over diagonals
        cvt(qb, abf, (long)Mq * DM);
        cvt(kb, bbf, (long)Mq * DM);
        launch_gemm(GF_BTRANS, abf, bbf, nullptr, nullptr, big, Lq, Lq, DM,
                    (long)Lq * DM, (long)Lq * DM, (long)Lq * Lq, BK, stream);
        diagmean_k<<<eg((long)BK * Lq), 256, 0, stream>>>(big, mv, BK, Lq);
        topk_k<<<dim3((BK + 63) / 64), 64, 0, stream>>>(mv, wgt, dly, BK, Lq);
        agg_k<<<eg((long)Mq * DM), 256, 0, stream>>>(vb, wgt, dly, ta, BK, Lq);
        cvt(ta, abf, (long)Mq * DM);
        cvt(Wo, bbf, (long)DM * DM);
        launch_gemm(GF_BIAS | GF_RESID, abf, bbf, bo, xq, dst, Mq, DM, DM, 0, 0, 0, 1, stream);
    };

    // circular conv1d: y (+)= sum_j shift(x, j-1) @ W[j]   W:[3,K,N]
    auto circ_conv = [&](const float* x, int L, int K, int N, const float* W,
                         float* y, int accum, float* shbuf) {
        int M = BK * L;
        shift_k<<<eg((long)M * K), 256, 0, stream>>>(x, shbuf, BK, L, K, -1);
        cvt(shbuf, abf, (long)M * K);
        cvt(W, bbf, (long)K * N);
        launch_gemm(accum ? GF_ACCUM : 0, abf, bbf, nullptr, nullptr, y, M, N, K, 0, 0, 0, 1, stream);
        cvt(x, abf, (long)M * K);
        cvt(W + (long)K * N, bbf, (long)K * N);
        launch_gemm(GF_ACCUM, abf, bbf, nullptr, nullptr, y, M, N, K, 0, 0, 0, 1, stream);
        shift_k<<<eg((long)M * K), 256, 0, stream>>>(x, shbuf, BK, L, K, +1);
        cvt(shbuf, abf, (long)M * K);
        cvt(W + 2 * (long)K * N, bbf, (long)K * N);
        launch_gemm(GF_ACCUM, abf, bbf, nullptr, nullptr, y, M, N, K, 0, 0, 0, 1, stream);
    };

    // ---------------- encoder ----------------
    circ_conv(xe, LE, CIN, DM, P(5), enc, 0, ta);
    for (int l = 0; l < 3; ++l) {
        int pb = 7 + l * 10;
        attn_block(enc, LE, enc, LE, pb, tb);
        decomp_k<<<eg(ME * DM), 256, 0, stream>>>(tb, enc, nullptr, BK, LE, DM, 0);
        cvt(enc, abf, ME * DM);
        cvt(P(pb + 8), bbf, (long)DM * DFF);
        launch_gemm(GF_GELU, abf, bbf, nullptr, nullptr, big, (int)ME, DFF, DM, 0, 0, 0, 1, stream);
        cvt(big, abf, ME * DFF);
        cvt(P(pb + 9), bbf, (long)DFF * DM);
        launch_gemm(GF_RESID, abf, bbf, nullptr, enc, tb, (int)ME, DM, DFF, 0, 0, 0, 1, stream);
        decomp_k<<<eg(ME * DM), 256, 0, stream>>>(tb, enc, nullptr, BK, LE, DM, 0);
    }
    ln_k<<<eg(ME), 256, 0, stream>>>(enc, enc, P(37), P(38), ME);
    subTmean_k<<<eg((long)BK * DM), 256, 0, stream>>>(enc, BK, LE);

    // ---------------- decoder init ----------------
    decomp_k<<<eg(ME * CIN), 256, 0, stream>>>(xe, sxe, txe, BK, LE, CIN, 1);
    meanT_k<<<eg((long)BK * CIN), 256, 0, stream>>>(xe, meanb, BK, LE, CIN);
    build_trend_k<<<eg(MD * CIN), 256, 0, stream>>>(txe, meanb, trend);
    build_seas_k<<<eg(MD * CIN), 256, 0, stream>>>(sxe, seasi);
    circ_conv(seasi, LD, CIN, DM, P(6), dec, 0, ta);

    // ---------------- decoder layers ----------------
    for (int l = 0; l < 2; ++l) {
        int pb = 39 + l * 19;
        attn_block(dec, LD, dec, LD, pb, tb);                                    // self
        decomp_k<<<eg(MD * DM), 256, 0, stream>>>(tb, dec, tt, BK, LD, DM, 1);   // t1
        attn_block(dec, LD, enc, LE, pb + 8, tb);                                // cross
        decomp_k<<<eg(MD * DM), 256, 0, stream>>>(tb, dec, tt, BK, LD, DM, 2);   // +t2
        cvt(dec, abf, MD * DM);
        cvt(P(pb + 16), bbf, (long)DM * DFF);
        launch_gemm(GF_GELU, abf, bbf, nullptr, nullptr, big, (int)MD, DFF, DM, 0, 0, 0, 1, stream);
        cvt(big, abf, MD * DFF);
        cvt(P(pb + 17), bbf, (long)DFF * DM);
        launch_gemm(GF_RESID, abf, bbf, nullptr, dec, tb, (int)MD, DM, DFF, 0, 0, 0, 1, stream);
        decomp_k<<<eg(MD * DM), 256, 0, stream>>>(tb, dec, tt, BK, LD, DM, 2);   // +t3
        circ_conv(tt, LD, DM, CIN, P(pb + 18), trend, 1, ta);                    // trend += proj
    }

    // ---------------- head ----------------
    ln_k<<<eg(MD), 256, 0, stream>>>(dec, dec, P(77), P(78), MD);
    subTmean_k<<<eg((long)BK * DM), 256, 0, stream>>>(dec, BK, LD);
    cvt(dec, abf, MD * DM);
    cvt(P(79), bbf, (long)DM * CIN);
    launch_gemm(GF_BIAS, abf, bbf, P(80), nullptr, s1, (int)MD, CIN, DM, 0, 0, 0, 1, stream);
    add2_k<<<eg(MD * CIN), 256, 0, stream>>>(trend, s1, ta, MD * CIN);
    cvt(ta, abf, MD * CIN);
    cvt(P(81), bbf, (long)CIN * CIN);
    launch_gemm(GF_BIAS, abf, bbf, P(82), nullptr, ofb, (int)MD, CIN, CIN, 0, 0, 0, 1, stream);
    slice_k<<<eg(ME * CIN), 256, 0, stream>>>(ofb, (float*)d_out);
}
